// TransformerDecoder_61048665145734
// MI455X (gfx1250) — matrix operand order repaired and checked by reading, not compiled, not run on hardware
//
#include <hip/hip_runtime.h>
#include <math.h>

typedef __bf16 bf16;
typedef __attribute__((ext_vector_type(8)))  bf16  v8bf;
typedef __attribute__((ext_vector_type(16))) bf16  v16bf;
typedef __attribute__((ext_vector_type(8)))  float v8f;

// ---------------------------------------------------------------------------
// helpers
// ---------------------------------------------------------------------------
static __device__ inline bf16 f2bf(float f) {
    unsigned u = __builtin_bit_cast(unsigned, f);
    unsigned r = (u + 0x7FFFu + ((u >> 16) & 1u)) >> 16;
    unsigned short s = (unsigned short)r;
    return __builtin_bit_cast(bf16, s);
}

// 16 contiguous bf16 -> v16bf (two b128 loads)
static __device__ inline v16bf load16(const bf16* p) {
    v8bf lo = *(const v8bf*)p;
    v8bf hi = *(const v8bf*)(p + 8);
    return __builtin_shufflevector(lo, hi, 0,1,2,3,4,5,6,7,8,9,10,11,12,13,14,15);
}

// A-matrix fragment (16x32 bf16) for one lane: row pointer `row` (this lane's M row),
// half = lane>>4.  Elements 0..15 <- K = half*16 + i
static __device__ inline v16bf load_a_frag(const bf16* row, int half) {
    v8bf lo = *(const v8bf*)(row + half * 16);
    v8bf hi = *(const v8bf*)(row + half * 16 + 8);
    return __builtin_shufflevector(lo, hi, 0,1,2,3,4,5,6,7,8,9,10,11,12,13,14,15);
}

static __device__ inline v8f wmma_bf16(v16bf a, v16bf b, v8f c) {
    return __builtin_amdgcn_wmma_f32_16x16x32_bf16(false, a, false, b, (short)0, c,
                                                   false, false);
}

static __device__ inline v8f zero8() {
    v8f c;
    #pragma unroll
    for (int i = 0; i < 8; ++i) c[i] = 0.0f;
    return c;
}

// ---------------------------------------------------------------------------
// weight convert + transpose : dst[n*K + k] = bf16(src[k*N + n]); batched over z
// ---------------------------------------------------------------------------
__global__ void cvt_transpose_kernel(const float* __restrict__ src,
                                     bf16* __restrict__ dst, int K, int N) {
    size_t total = (size_t)K * N;
    size_t idx = (size_t)blockIdx.x * blockDim.x + threadIdx.x;
    if (idx >= total) return;
    size_t base = (size_t)blockIdx.z * total;
    int n = (int)(idx % N);
    int k = (int)(idx / N);
    dst[base + (size_t)n * K + k] = f2bf(src[base + idx]);
}

// ---------------------------------------------------------------------------
// embedding + sinusoidal positional encoding.  One block per token.
// ---------------------------------------------------------------------------
__global__ void embed_kernel(const int* __restrict__ tok,
                             const float* __restrict__ emb,
                             float* __restrict__ x, bf16* __restrict__ xb,
                             int S, int D) {
    int t = blockIdx.x;          // 0 .. B*S-1
    int s = t % S;
    int id = tok[t];
    float sc = sqrtf((float)D);
    float c0 = logf(10000.0f) / (float)D;
    for (int d = threadIdx.x; d < D; d += blockDim.x) {
        float e = emb[(size_t)id * D + d] * sc;
        float freq = __expf(-(float)(d & ~1) * c0);
        float ang = (float)s * freq;
        float pe = (d & 1) ? __cosf(ang) : __sinf(ang);
        float v = e + pe;
        x[(size_t)t * D + d] = v;
        xb[(size_t)t * D + d] = f2bf(v);
    }
}

// ---------------------------------------------------------------------------
// generic bf16 WMMA GEMM:  out[M,N] = A[M,K] @ Bt[N,K]^T + bias, opt ReLU.
// 256 threads = 8 waves.  Each wave owns a 16(M) x 64(N) output tile
// (4 accumulators sharing one A fragment), software-pipelined one k-step
// ahead so WMMAs never wait on same-iteration loads.
// ---------------------------------------------------------------------------
__global__ void gemm_bf16_kernel(const bf16* __restrict__ A,
                                 const bf16* __restrict__ Bt,
                                 const float* __restrict__ bias,
                                 float* __restrict__ outF,
                                 bf16* __restrict__ outB,
                                 int M, int N, int K, int relu) {
    int lane = threadIdx.x & 31;
    int wave = threadIdx.x >> 5;            // 0..7  -> M strip
    int row0 = (blockIdx.y * 8 + wave) * 16;
    int col0 = blockIdx.x * 64;
    if (row0 >= M || col0 >= N) return;

    int half = lane >> 4, r = lane & 15;
    const bf16* a_row = A  + (size_t)(row0 + r) * K;
    const bf16* b_row = Bt + (size_t)(col0 + r) * K;   // j-th tile at +j*16*K

    v8f c[4];
    #pragma unroll
    for (int j = 0; j < 4; ++j) c[j] = zero8();

    // prologue: fragments for k0 = 0
    v16bf a = load_a_frag(a_row, half);
    v16bf b[4];
    #pragma unroll
    for (int j = 0; j < 4; ++j)
        b[j] = load16(b_row + (size_t)j * 16 * K + half * 16);

    // steady state: issue next k-step's loads, then current k-step's WMMAs
    for (int k0 = 0; k0 + 32 < K; k0 += 32) {
        int kn = k0 + 32;
        if (kn + 64 < K) __builtin_prefetch(a_row + kn + 64, 0, 1);
        v16bf an = load_a_frag(a_row + kn, half);
        v16bf bn[4];
        #pragma unroll
        for (int j = 0; j < 4; ++j)
            bn[j] = load16(b_row + (size_t)j * 16 * K + kn + half * 16);
        #pragma unroll
        for (int j = 0; j < 4; ++j) c[j] = wmma_bf16(a, b[j], c[j]);
        a = an;
        #pragma unroll
        for (int j = 0; j < 4; ++j) b[j] = bn[j];
    }
    // epilogue: last k-step
    #pragma unroll
    for (int j = 0; j < 4; ++j) c[j] = wmma_bf16(a, b[j], c[j]);

    int m0 = row0 + half * 8;
    #pragma unroll
    for (int j = 0; j < 4; ++j) {
        int col = col0 + j * 16 + r;
        float bv = bias ? bias[col] : 0.0f;
        #pragma unroll
        for (int i = 0; i < 8; ++i) {
            float v = c[j][i] + bv;
            if (relu) v = fmaxf(v, 0.0f);
            size_t idx = (size_t)(m0 + i) * N + col;
            if (outF) outF[idx] = v;
            if (outB) outB[idx] = f2bf(v);
        }
    }
}

// ---------------------------------------------------------------------------
// V transpose: v[b*S+t][h*64+j] -> vT[((b*H+h)*64+j)*S + t]
// ---------------------------------------------------------------------------
__global__ void v_transpose_kernel(const bf16* __restrict__ v,
                                   bf16* __restrict__ vT,
                                   int B, int S, int D, int H) {
    size_t total = (size_t)B * S * D;
    size_t idx = (size_t)blockIdx.x * blockDim.x + threadIdx.x;
    if (idx >= total) return;
    int d = (int)(idx % D);
    int t = (int)((idx / D) % S);
    int b = (int)(idx / ((size_t)D * S));
    int h = d >> 6, j = d & 63;
    vT[(((size_t)b * H + h) * 64 + j) * S + t] = v[idx];
}

// ---------------------------------------------------------------------------
// flash-style causal attention.  blockDim = 32 (one wave).
// grid = (S/16, H, B).  Q,K : [B*S, D] bf16 ;  vT : [B,H,64,S] bf16.
// ---------------------------------------------------------------------------
__global__ void attention_kernel(const bf16* __restrict__ q,
                                 const bf16* __restrict__ k,
                                 const bf16* __restrict__ vT,
                                 bf16* __restrict__ attn,
                                 int B, int S, int H) {
    const int D = 512, dk = 64;
    const float NEG_INF = -__builtin_inff();
    int lane = threadIdx.x & 31;
    int half = lane >> 4, r = lane & 15;
    int qt = blockIdx.x, h = blockIdx.y, b = blockIdx.z;
    int t0 = qt * 16;

    const bf16* qbase = q + (size_t)b * S * D;
    const bf16* kbase = k + (size_t)b * S * D;
    const bf16* vtb   = vT + ((size_t)b * H + h) * dk * S;

    // Q fragments for feature halves kk = 0, 1 (features kk*32 .. +31)
    v16bf qa[2];
    #pragma unroll
    for (int kk = 0; kk < 2; ++kk)
        qa[kk] = load_a_frag(qbase + (size_t)(t0 + r) * D + h * dk + kk * 32, half);

    v8f o[4];
    #pragma unroll
    for (int f = 0; f < 4; ++f) o[f] = zero8();

    __shared__ float Mrow[16], Lrow[16];
    __shared__ __align__(16) bf16 P[16][32];
    if (threadIdx.x < 16) { Mrow[threadIdx.x] = NEG_INF; Lrow[threadIdx.x] = 0.0f; }
    __syncthreads();

    int nkb = (t0 + 15) / 32 + 1;     // causal: key blocks of 32
    for (int kb = 0; kb < nkb; ++kb) {
        int kt = kb * 32;
        // scores: two 16-column tiles of keys
        v8f sc[2];
        #pragma unroll
        for (int ct = 0; ct < 2; ++ct) {
            sc[ct] = zero8();
            #pragma unroll
            for (int kk = 0; kk < 2; ++kk) {
                const bf16* p = kbase + (size_t)(kt + ct * 16 + r) * D +
                                h * dk + kk * 32 + half * 16;
                sc[ct] = wmma_bf16(qa[kk], load16(p), sc[ct]);
            }
        }
        // scale (1/sqrt(64)) + causal mask. element (i): row m=i+8*half, col=r
        #pragma unroll
        for (int ct = 0; ct < 2; ++ct)
            #pragma unroll
            for (int i = 0; i < 8; ++i) {
                float s = sc[ct][i] * 0.125f;
                if (kt + ct * 16 + r > t0 + i + 8 * half) s = NEG_INF;
                sc[ct][i] = s;
            }

        float alpha[8], newM[8], rs[8];
        #pragma unroll
        for (int i = 0; i < 8; ++i) {
            float mx = fmaxf(sc[0][i], sc[1][i]);
            for (int msk = 1; msk < 16; msk <<= 1)
                mx = fmaxf(mx, __shfl_xor(mx, msk, 32));
            int m = i + 8 * half;
            float om = Mrow[m];
            float nm = fmaxf(om, mx);
            newM[i] = nm;
            alpha[i] = (om > NEG_INF) ? __expf(om - nm) : 0.0f;
            float rsum = 0.0f;
            #pragma unroll
            for (int ct = 0; ct < 2; ++ct) {
                float s = sc[ct][i];
                float pv = (s > NEG_INF) ? __expf(s - nm) : 0.0f;
                P[m][ct * 16 + r] = f2bf(pv);
                rsum += pv;
            }
            for (int msk = 1; msk < 16; msk <<= 1)
                rsum += __shfl_xor(rsum, msk, 32);
            rs[i] = rsum;
        }
        if (r == 0) {      // lane 0 -> rows 0..7, lane 16 -> rows 8..15
            #pragma unroll
            for (int i = 0; i < 8; ++i) {
                int m = i + 8 * half;
                Mrow[m] = newM[i];
                Lrow[m] = Lrow[m] * alpha[i] + rs[i];
            }
        }
        __syncthreads();   // P + stats visible

        #pragma unroll
        for (int f = 0; f < 4; ++f)
            #pragma unroll
            for (int i = 0; i < 8; ++i) o[f][i] *= alpha[i];

        // P (16x32) as A fragment from LDS; V block as B fragments from vT
        v16bf pa = load_a_frag(&P[r][0], half);
        #pragma unroll
        for (int f = 0; f < 4; ++f) {
            const bf16* vp = vtb + (size_t)(f * 16 + r) * S + kt + half * 16;
            o[f] = wmma_bf16(pa, load16(vp), o[f]);
        }
        __syncthreads();   // done with P before next overwrite
    }

    // normalize rows and store bf16
    #pragma unroll
    for (int f = 0; f < 4; ++f)
        #pragma unroll
        for (int i = 0; i < 8; ++i) {
            int m = i + 8 * half;
            float l = Lrow[m];
            float val = (l > 0.0f) ? o[f][i] / l : 0.0f;
            attn[(size_t)(b * S + t0 + m) * 512 + h * dk + f * 16 + r] = f2bf(val);
        }
}

// ---------------------------------------------------------------------------
// residual add + LayerNorm.  One block (256 thr) per token, D = 512.
// ---------------------------------------------------------------------------
__global__ void add_ln_kernel(const float* __restrict__ x,
                              const float* __restrict__ y,
                              const float* __restrict__ g,
                              const float* __restrict__ bt,
                              float* __restrict__ outF,
                              bf16* __restrict__ outB, int D) {
    int t = blockIdx.x;
    int tid = threadIdx.x;
    const int PER = 2;               // 512 / 256
    float rv[PER];
    float s = 0.0f, s2 = 0.0f;
    #pragma unroll
    for (int i = 0; i < PER; ++i) {
        int d = tid + i * 256;
        float v = x[(size_t)t * D + d];
        if (y) v += y[(size_t)t * D + d];
        rv[i] = v; s += v; s2 += v * v;
    }
    for (int m = 16; m >= 1; m >>= 1) {
        s  += __shfl_xor(s,  m, 32);
        s2 += __shfl_xor(s2, m, 32);
    }
    __shared__ float ws1[8], ws2[8];
    __shared__ float mu_s, rstd_s;
    int wv = tid >> 5, ln = tid & 31;
    if (ln == 0) { ws1[wv] = s; ws2[wv] = s2; }
    __syncthreads();
    if (tid == 0) {
        float a = 0.0f, b2 = 0.0f;
        for (int i = 0; i < 8; ++i) { a += ws1[i]; b2 += ws2[i]; }
        float mu = a / (float)D;
        float var = b2 / (float)D - mu * mu;
        mu_s = mu;
        rstd_s = rsqrtf(var + 1e-5f);
    }
    __syncthreads();
    float mu = mu_s, rstd = rstd_s;
    #pragma unroll
    for (int i = 0; i < PER; ++i) {
        int d = tid + i * 256;
        float o = (rv[i] - mu) * rstd * g[d] + bt[d];
        if (outF) outF[(size_t)t * D + d] = o;
        if (outB) outB[(size_t)t * D + d] = f2bf(o);
    }
}

// ---------------------------------------------------------------------------
// host side
// ---------------------------------------------------------------------------
static inline char* carve(char*& p, size_t bytes) {
    char* r = p;
    p += (bytes + 255) & ~(size_t)255;
    return r;
}

extern "C" void kernel_launch(void* const* d_in, const int* in_sizes, int n_in,
                              void* d_out, int out_size, void* d_ws, size_t ws_size,
                              hipStream_t stream) {
    (void)n_in; (void)out_size; (void)ws_size;
    const int S = 2048, D = 512, H = 8, L = 6, F = 2048, V = 32000;
    const int B = in_sizes[0] / S;
    const int M = B * S;

    const int*   tok  = (const int*)  d_in[0];
    const float* emb  = (const float*)d_in[1];
    const float* Wq   = (const float*)d_in[2];
    const float* bq   = (const float*)d_in[3];
    const float* Wk   = (const float*)d_in[4];
    const float* bk   = (const float*)d_in[5];
    const float* Wv   = (const float*)d_in[6];
    const float* bv   = (const float*)d_in[7];
    const float* Wo   = (const float*)d_in[8];
    const float* bo   = (const float*)d_in[9];
    const float* W1   = (const float*)d_in[10];
    const float* b1   = (const float*)d_in[11];
    const float* W2   = (const float*)d_in[12];
    const float* b2   = (const float*)d_in[13];
    const float* ln1g = (const float*)d_in[14];
    const float* ln1b = (const float*)d_in[15];
    const float* ln2g = (const float*)d_in[16];
    const float* ln2b = (const float*)d_in[17];
    const float* fng  = (const float*)d_in[18];
    const float* fnb  = (const float*)d_in[19];
    const float* Wout = (const float*)d_in[20];
    const float* bout = (const float*)d_in[21];

    char* p = (char*)d_ws;
    float* x     = (float*)carve(p, (size_t)M * D * 4);
    bf16*  xb    = (bf16*) carve(p, (size_t)M * D * 2);
    bf16*  qb    = (bf16*) carve(p, (size_t)M * D * 2);
    bf16*  kb    = (bf16*) carve(p, (size_t)M * D * 2);
    bf16*  vb    = (bf16*) carve(p, (size_t)M * D * 2);
    bf16*  vT    = (bf16*) carve(p, (size_t)M * D * 2);
    bf16*  attnb = (bf16*) carve(p, (size_t)M * D * 2);
    bf16*  hb    = (bf16*) carve(p, (size_t)M * F * 2);
    float* tmpF  = (float*)carve(p, (size_t)M * D * 4);
    bf16*  Wqt   = (bf16*) carve(p, (size_t)L * D * D * 2);
    bf16*  Wkt   = (bf16*) carve(p, (size_t)L * D * D * 2);
    bf16*  Wvt   = (bf16*) carve(p, (size_t)L * D * D * 2);
    bf16*  Wot   = (bf16*) carve(p, (size_t)L * D * D * 2);
    bf16*  W1t   = (bf16*) carve(p, (size_t)L * D * F * 2);
    bf16*  W2t   = (bf16*) carve(p, (size_t)L * D * F * 2);
    bf16*  WoutT = (bf16*) carve(p, (size_t)V * D * 2);

    dim3 blk(256);
    // weight conversion + transpose (bf16, N-major)
    cvt_transpose_kernel<<<dim3((D * D + 255) / 256, 1, L), blk, 0, stream>>>(Wq, Wqt, D, D);
    cvt_transpose_kernel<<<dim3((D * D + 255) / 256, 1, L), blk, 0, stream>>>(Wk, Wkt, D, D);
    cvt_transpose_kernel<<<dim3((D * D + 255) / 256, 1, L), blk, 0, stream>>>(Wv, Wvt, D, D);
    cvt_transpose_kernel<<<dim3((D * D + 255) / 256, 1, L), blk, 0, stream>>>(Wo, Wot, D, D);
    cvt_transpose_kernel<<<dim3((D * F + 255) / 256, 1, L), blk, 0, stream>>>(W1, W1t, D, F);
    cvt_transpose_kernel<<<dim3((D * F + 255) / 256, 1, L), blk, 0, stream>>>(W2, W2t, F, D);
    cvt_transpose_kernel<<<dim3((int)(((size_t)D * V + 255) / 256), 1, 1), blk, 0, stream>>>(Wout, WoutT, D, V);

    embed_kernel<<<M, blk, 0, stream>>>(tok, emb, x, xb, S, D);

    dim3 gD(D / 64, M / 128);      // N=512 GEMMs
    dim3 gF(F / 64, M / 128);      // N=2048 GEMM
    for (int l = 0; l < L; ++l) {
        const bf16* wq = Wqt + (size_t)l * D * D;
        const bf16* wk = Wkt + (size_t)l * D * D;
        const bf16* wv = Wvt + (size_t)l * D * D;
        const bf16* wo = Wot + (size_t)l * D * D;
        const bf16* w1 = W1t + (size_t)l * D * F;
        const bf16* w2 = W2t + (size_t)l * D * F;

        gemm_bf16_kernel<<<gD, blk, 0, stream>>>(xb, wq, bq + l * D, nullptr, qb, M, D, D, 0);
        gemm_bf16_kernel<<<gD, blk, 0, stream>>>(xb, wk, bk + l * D, nullptr, kb, M, D, D, 0);
        gemm_bf16_kernel<<<gD, blk, 0, stream>>>(xb, wv, bv + l * D, nullptr, vb, M, D, D, 0);
        v_transpose_kernel<<<(int)(((size_t)M * D + 255) / 256), blk, 0, stream>>>(vb, vT, B, S, D, H);
        attention_kernel<<<dim3(S / 16, H, B), dim3(32), 0, stream>>>(qb, kb, vT, attnb, B, S, H);
        gemm_bf16_kernel<<<gD, blk, 0, stream>>>(attnb, wo, bo + l * D, tmpF, nullptr, M, D, D, 0);
        add_ln_kernel<<<M, blk, 0, stream>>>(x, tmpF, ln1g + l * D, ln1b + l * D, x, xb, D);
        gemm_bf16_kernel<<<gF, blk, 0, stream>>>(xb, w1, b1 + l * F, nullptr, hb, M, F, D, 1);
        gemm_bf16_kernel<<<gD, blk, 0, stream>>>(hb, w2, b2 + l * D, tmpF, nullptr, M, D, F, 0);
        add_ln_kernel<<<M, blk, 0, stream>>>(x, tmpF, ln2g + l * D, ln2b + l * D, x, xb, D);
    }

    // final LN + vocab projection into d_out (fp32)
    add_ln_kernel<<<M, blk, 0, stream>>>(x, nullptr, fng, fnb, nullptr, xb, D);
    gemm_bf16_kernel<<<dim3(V / 64, M / 128), blk, 0, stream>>>(xb, WoutT, bout,
                                                               (float*)d_out, nullptr,
                                                               M, V, D, 0);
}
